// MultilayerLSTMScan_74500502716488
// MI455X (gfx1250) — matrix operand order repaired and checked by reading, not compiled, not run on hardware
//
#include <hip/hip_runtime.h>

#define L_LAYERS 2
#define B_ 64
#define T_ 512
#define H_ 1024
#define G4H_ 4096

typedef __attribute__((ext_vector_type(16))) __bf16 v16bf;
typedef __attribute__((ext_vector_type(8)))  float  v8f;

union Frag { v16bf v; uint4 q[2]; };

__device__ __forceinline__ unsigned short f2bf(float f) {
  unsigned u = __float_as_uint(f);
  unsigned r = u + 0x7FFFu + ((u >> 16) & 1u);   // round-to-nearest-even
  return (unsigned short)(r >> 16);
}

// branch-free fast activations on native v_exp_f32
__device__ __forceinline__ float fast_sigmoid(float x) {
  return 1.f / (1.f + __expf(-x));
}
__device__ __forceinline__ float fast_tanh(float x) {
  return 1.f - 2.f / (1.f + __expf(2.f * x));
}

// ---------------------------------------------------------------------------
// One-time prep: pack [Wi;Wh] (K=2048 x N=4096, fp32) into WMMA B-fragment
// layout, bf16. Layout: [l][jtile 64][gate 4][kstep 64][lane 32][16 bf16].
// lane<16 : column n = base + lane,    k = 32*ks + j16       (K 0..15)
// lane>=16: column n = base + lane-16, k = 32*ks + 16 + j16  (K 16..31)
// ---------------------------------------------------------------------------
__global__ void pack_weights_kernel(const float* __restrict__ Wi,
                                    const float* __restrict__ Wh,
                                    unsigned short* __restrict__ wpack) {
  long long e = (long long)blockIdx.x * blockDim.x + threadIdx.x;
  const long long PER_L = 8388608LL;               // 64*4*64*32*16
  if (e >= 2 * PER_L) return;
  int l = (int)(e / PER_L);
  unsigned r = (unsigned)(e % PER_L);
  int j16   = r & 15;
  int lane  = (r >> 4) & 31;
  int ks    = (r >> 9) & 63;
  int gate  = (r >> 15) & 3;
  int jtile = (r >> 17) & 63;
  int n = gate * H_ + jtile * 16 + (lane & 15);
  int k = ks * 32 + ((lane < 16) ? j16 : 16 + j16);
  float v = (k < H_) ? Wi[((long long)l * H_ + k) * G4H_ + n]
                     : Wh[((long long)l * H_ + (k - H_)) * G4H_ + n];
  wpack[e] = f2bf(v);
}

__global__ void convert_x_kernel(const float* __restrict__ x,
                                 unsigned short* __restrict__ xb, int n) {
  int i = blockIdx.x * blockDim.x + threadIdx.x;
  if (i < n) xb[i] = f2bf(x[i]);
}

__global__ void init_state_kernel(const float* __restrict__ c0,
                                  const float* __restrict__ h0,
                                  const float* __restrict__ mask,
                                  float* __restrict__ c, float* __restrict__ h,
                                  unsigned short* __restrict__ hm0, int n) {
  int i = blockIdx.x * blockDim.x + threadIdx.x;
  if (i < n) {
    float hv = h0[i];
    c[i] = c0[i];
    h[i] = hv;
    hm0[i] = f2bf(hv * mask[i]);   // parity-0 masked-h for t=0, both layers
  }
}

// ---------------------------------------------------------------------------
// One LSTM timestep for one layer.
// grid = 64 blocks (jtile = 16-column slice of H), block = 512 = 16 waves.
// wave = (m-tile in 0..3) x (gate in 0..3). K = 2048 = [input 1024 | hm 1024].
// K-loop is split into the two operand halves so each half has a single base
// pointer and the compiler folds k-steps into immediate load offsets.
// ---------------------------------------------------------------------------
__global__ __launch_bounds__(512)
void lstm_step_kernel(const unsigned short* __restrict__ in, long long in_stride,
                      long long in_off,
                      const unsigned short* __restrict__ wpack_l,
                      const unsigned short* __restrict__ hm_in,
                      unsigned short* __restrict__ hm_out,
                      float* __restrict__ h_l, float* __restrict__ c_l,
                      const float* __restrict__ bh_l,
                      const float* __restrict__ mask_l,
                      const int* __restrict__ lengths, int t,
                      unsigned short* __restrict__ ybf,
                      float* __restrict__ yf32) {
  __shared__ float lds[16 * 256];                  // [gate*4+m][256]
  const int tid  = threadIdx.x;
  const int lane = tid & 31;
  const int wave = tid >> 5;
  const int m    = wave & 3;
  const int gate = wave >> 2;
  const int jtile = blockIdx.x;
  const int row = m * 16 + (lane & 15);            // batch row for A fragment
  const int sel = (lane & 16) ? 16 : 0;            // A K-offset select per half-wave

  const unsigned short* arow0 = in + (long long)row * in_stride + in_off + sel;
  const unsigned short* arow1 = hm_in + (long long)row * H_ + sel;
  const unsigned short* wp =
      wpack_l + ((((long long)jtile * 4 + gate) * 64) * 32 + lane) * 16;

  v8f acc = {};
  // ---- first half: input operand (K = 0..1023) ----
#pragma unroll 4
  for (int ks = 0; ks < 32; ++ks) {
    Frag a, b;
    const unsigned short* ab = arow0 + ks * 32;
    a.q[0] = *(const uint4*)(ab);
    a.q[1] = *(const uint4*)(ab + 8);
    const unsigned short* bb = wp + ks * 512;
    b.q[0] = *(const uint4*)(bb);
    b.q[1] = *(const uint4*)(bb + 8);
    acc = __builtin_amdgcn_wmma_f32_16x16x32_bf16(
        false, a.v, false, b.v, (short)0, acc, false, false);
  }
  // ---- second half: recurrent operand h*mask (K = 1024..2047) ----
#pragma unroll 4
  for (int ks = 0; ks < 32; ++ks) {
    Frag a, b;
    const unsigned short* ab = arow1 + ks * 32;
    a.q[0] = *(const uint4*)(ab);
    a.q[1] = *(const uint4*)(ab + 8);
    const unsigned short* bb = wp + (ks + 32) * 512;
    b.q[0] = *(const uint4*)(bb);
    b.q[1] = *(const uint4*)(bb + 8);
    acc = __builtin_amdgcn_wmma_f32_16x16x32_bf16(
        false, a.v, false, b.v, (short)0, acc, false, false);
  }

  // spill accumulator tile to LDS: C/D layout lane<16 -> M=r, lane>=16 -> M=8+r
  {
    const int base = (gate * 4 + m) * 256;
    const int mrow_off = (lane >> 4) << 3;
    const int col = lane & 15;
#pragma unroll
    for (int r2 = 0; r2 < 8; ++r2)
      lds[base + (r2 + mrow_off) * 16 + col] = acc[r2];
  }
  __syncthreads();

  // pointwise LSTM cell: 1024 (b,j) elements in this block, 2 per thread
#pragma unroll
  for (int p = 0; p < 2; ++p) {
    int e = tid + p * 512;
    int mm  = e >> 8;
    int idx = e & 255;
    int b_row = mm * 16 + (idx >> 4);
    int j = jtile * 16 + (idx & 15);
    float zi = lds[(0 * 4 + mm) * 256 + idx] + bh_l[j];
    float zf = lds[(1 * 4 + mm) * 256 + idx] + bh_l[H_ + j];
    float zg = lds[(2 * 4 + mm) * 256 + idx] + bh_l[2 * H_ + j];
    float zo = lds[(3 * 4 + mm) * 256 + idx] + bh_l[3 * H_ + j];
    float ig = fast_sigmoid(zi);
    float fg = fast_sigmoid(zf);
    float og = fast_sigmoid(zo);
    float gg = fast_tanh(zg);
    long long sidx = (long long)b_row * H_ + j;
    float cold = c_l[sidx], hold = h_l[sidx];
    float nc = fg * cold + ig * gg;
    float nh = og * fast_tanh(nc);
    bool keep = t < lengths[b_row];
    float cN = keep ? nc : cold;
    float hN = keep ? nh : hold;
    c_l[sidx] = cN;
    h_l[sidx] = hN;
    hm_out[sidx] = f2bf(hN * mask_l[sidx]);        // recurrent-dropout masked h
    if (ybf) ybf[sidx] = f2bf(nh);                 // layer output (unmasked) -> next layer
    else     yf32[((long long)b_row * T_ + t) * H_ + j] = nh;   // final output
  }
}

extern "C" void kernel_launch(void* const* d_in, const int* in_sizes, int n_in,
                              void* d_out, int out_size, void* d_ws, size_t ws_size,
                              hipStream_t stream) {
  const float* x       = (const float*)d_in[0];
  const int*   lengths = (const int*)d_in[1];
  const float* h_mask  = (const float*)d_in[2];
  const float* Wi      = (const float*)d_in[3];
  const float* Wh      = (const float*)d_in[4];
  const float* bh      = (const float*)d_in[5];
  const float* c0      = (const float*)d_in[6];
  const float* h0      = (const float*)d_in[7];
  float* out = (float*)d_out;
  char* ws = (char*)d_ws;

  // workspace layout (bytes)
  unsigned short* xb    = (unsigned short*)(ws);                   // 64 MB  x bf16
  unsigned short* wpack = (unsigned short*)(ws + 67108864LL);      // 32 MB  packed [Wi;Wh]
  unsigned short* hm    = (unsigned short*)(ws + 100663296LL);     // [2 parity][2 layer][65536] u16
  unsigned short* y0    = (unsigned short*)(ws + 101187584LL);     // 128 KB layer-0 output bf16
  float* hstate = (float*)(ws + 101318656LL);                      // [2][65536] f32
  float* cstate = (float*)(ws + 101842944LL);                      // [2][65536] f32
  (void)in_sizes; (void)n_in; (void)out_size; (void)ws_size;

  const int NSTATE = L_LAYERS * B_ * H_;       // 131072
  const int NX     = B_ * T_ * H_;             // 33554432

  pack_weights_kernel<<<65536, 256, 0, stream>>>(Wi, Wh, wpack);
  convert_x_kernel<<<NX / 256, 256, 0, stream>>>(x, xb, NX);
  init_state_kernel<<<NSTATE / 256, 256, 0, stream>>>(c0, h0, h_mask,
                                                      cstate, hstate, hm, NSTATE);

  const long long PER_L = 8388608LL;           // packed u16 per layer
  for (int t = 0; t < T_; ++t) {
    int rp = t & 1, wp = rp ^ 1;
    // layer 0: input = x[:, t, :] (bf16), recurrent hm[rp][0] -> hm[wp][0]
    lstm_step_kernel<<<64, 512, 0, stream>>>(
        xb, (long long)T_ * H_, (long long)t * H_,
        wpack,
        hm + ((long long)rp * 2 + 0) * 65536,
        hm + ((long long)wp * 2 + 0) * 65536,
        hstate, cstate, bh, h_mask, lengths, t,
        y0, nullptr);
    // layer 1: input = y0 (layer-0 nh, unmasked), writes final output fp32
    lstm_step_kernel<<<64, 512, 0, stream>>>(
        y0, (long long)H_, 0LL,
        wpack + PER_L,
        hm + ((long long)rp * 2 + 1) * 65536,
        hm + ((long long)wp * 2 + 1) * 65536,
        hstate + 65536, cstate + 65536, bh + G4H_, h_mask + 65536, lengths, t,
        nullptr, out);
  }
}
